// PrDiMPSteepestDescentNewton_53128745452220
// MI455X (gfx1250) — matrix operand order repaired and checked by reading, not compiled, not run on hardware
//
#include <hip/hip_runtime.h>
#include <hip/hip_bf16.h>

typedef __bf16 bf16;
typedef __attribute__((ext_vector_type(16))) __bf16 v16bf;
typedef __attribute__((ext_vector_type(8)))  float  v8f;

union BFrag {
  v16bf v;
  float4 q[2];
  bf16   e[16];
};

#define I_N   3
#define S_N   32
#define C_N   512
#define H_N   22
#define W_N   22
#define P_N   484      // 22*22
#define PP_N  496      // padded to 31*16
#define HO    23
#define WO    23
#define OUT_N 529      // 23*23
#define T_N   16       // 4x4 taps
#define FLT_N 8192     // C_N * 16

__device__ inline float blockReduceSum(float v, float* sRed) {
  __syncthreads();
  for (int o = 16; o > 0; o >>= 1) v += __shfl_xor(v, o, 32);
  int w = threadIdx.x >> 5, ln = threadIdx.x & 31;
  if (ln == 0) sRed[w] = v;
  __syncthreads();
  if (w == 0) {
    float r = (ln < ((int)blockDim.x >> 5)) ? sRed[ln] : 0.f;
    for (int o = 4; o > 0; o >>= 1) r += __shfl_xor(r, o, 32);
    if (ln == 0) sRed[0] = r;
  }
  __syncthreads();
  return sRed[0];
}

__device__ inline float blockReduceMax(float v, float* sRed) {
  __syncthreads();
  for (int o = 16; o > 0; o >>= 1) v = fmaxf(v, __shfl_xor(v, o, 32));
  int w = threadIdx.x >> 5, ln = threadIdx.x & 31;
  if (ln == 0) sRed[w] = v;
  __syncthreads();
  if (w == 0) {
    float r = (ln < ((int)blockDim.x >> 5)) ? sRed[ln] : -3.0e38f;
    for (int o = 4; o > 0; o >>= 1) r = fmaxf(r, __shfl_xor(r, o, 32));
    if (ln == 0) sRed[0] = r;
  }
  __syncthreads();
  return sRed[0];
}

// ---------------------------------------------------------------------------
// One-time pack: feat fp32 -> bf16 in two layouts, filter copy, label density.
// grid: 96 blocks = (i,s) pairs, 256 threads.
// ---------------------------------------------------------------------------
__global__ void k_pack(const float* __restrict__ filt, const float* __restrict__ feat,
                       const float* __restrict__ bb, const float* __restrict__ lsl,
                       const float* __restrict__ freg,
                       float* __restrict__ fW, bf16* __restrict__ fbf,
                       bf16* __restrict__ featP, bf16* __restrict__ featT,
                       float* __restrict__ labelWS, float* __restrict__ scal) {
  __shared__ float sRed[32];
  int b = blockIdx.x;
  int i = b >> 5, s = b & 31;
  int tid = threadIdx.x;
  const float* fsrc = feat + (size_t)b * C_N * P_N;
  bf16* fp = featP + (size_t)b * PP_N * C_N;   // position-major [p][c]
  bf16* ft = featT + (size_t)b * C_N * PP_N;   // channel-major  [c][p]
  for (int idx = tid; idx < C_N * PP_N; idx += blockDim.x) {
    int c = idx / PP_N, p = idx - c * PP_N;
    float v = (p < P_N) ? fsrc[(size_t)c * P_N + p] : 0.f;
    bf16 bv = (bf16)v;
    ft[(size_t)c * PP_N + p] = bv;
    fp[(size_t)p * C_N + c] = bv;
  }
  if (i == 0) {
    for (int idx = tid; idx < FLT_N; idx += blockDim.x) {
      float v = filt[(size_t)s * FLT_N + idx];
      fW[(size_t)s * FLT_N + idx] = v;
      int c = idx >> 4, t = idx & 15;
      fbf[(size_t)s * FLT_N + t * C_N + c] = (bf16)v;   // tap-major bf16
    }
  }
  // label density (sigma=1, thresh=0, fh,fw even -> offset 0)
  const float* bp = bb + (size_t)b * 4;
  float c0 = (bp[1] + 0.5f * bp[3]) * (1.0f / 16.0f);   // row (y)
  float c1 = (bp[0] + 0.5f * bp[2]) * (1.0f / 16.0f);   // col (x)
  float* lab = labelWS + (size_t)b * OUT_N;
  float part = 0.f;
  for (int idx = tid; idx < OUT_N; idx += blockDim.x) {
    int y = idx / WO, x = idx - y * WO;
    float d0 = (float)y - c0, d1 = (float)x - c1;
    float g = expf(-0.5f * (d0 * d0 + d1 * d1)) * 0.15915494309189535f; // 1/(2*pi)
    lab[idx] = g;
    part += g;
  }
  float sum = blockReduceSum(part, sRed);
  float inv = 1.f / (sum + 1e-8f);
  for (int idx = tid; idx < OUT_N; idx += blockDim.x) lab[idx] *= inv;
  if (b == 0 && tid == 0) {
    float fr = freg[0];
    scal[0] = fmaxf(fr * fr, 1e-6f);  // reg
    scal[1] = expf(lsl[0]);           // step factor
  }
}

// ---------------------------------------------------------------------------
// Grouped 4x4 conv as WMMA GEMM: P[p,t]=sum_c feat[p,c]*B[c,t], then shifted
// gather to 23x23 scores. grid: 96 blocks = (i,s), 256 threads = 8 waves.
// Each wave owns 4 M-tiles sharing one B fragment; K-loop double-buffered.
// ---------------------------------------------------------------------------
__global__ void k_conv(const bf16* __restrict__ featP, const bf16* __restrict__ Bsrc,
                       float* __restrict__ outWS) {
  __shared__ float sP[PP_N * T_N];   // 31 KB tap-partials
  __shared__ bf16  sFB[FLT_N];       // 16 KB filter (tap-major)
  int pair = blockIdx.x;
  int s = pair & 31;
  int tid = threadIdx.x;
  int wave = tid >> 5, lane = tid & 31;
  int sel = lane >> 4, row = lane & 15;

  const unsigned int* src = (const unsigned int*)(Bsrc + (size_t)s * FLT_N);
  unsigned int* dst = (unsigned int*)sFB;
  for (int idx = tid; idx < FLT_N / 2; idx += blockDim.x) dst[idx] = src[idx];
  __syncthreads();

  const bf16* Abase = featP + (size_t)pair * PP_N * C_N;
  const bf16* brow = sFB + row * C_N;               // B lane column = tap
  const bf16* arow[4];
  int mt[4];
#pragma unroll
  for (int u = 0; u < 4; ++u) {
    mt[u] = wave + u * 8;
    int mc = mt[u] < 31 ? mt[u] : 30;               // clamp: tile 31 is a dummy
    arow[u] = Abase + (size_t)(mc * 16 + row) * C_N;
  }

  v8f acc[4] = {};
  BFrag aF[2][4], bF[2];
  auto loadK = [&](int kk, int buf) {
    int q0 = kk * 32;
    const bf16* bp = brow + q0 + sel * 16;
    bF[buf].q[0] = *(const float4*)(bp);
    bF[buf].q[1] = *(const float4*)(bp + 8);
#pragma unroll
    for (int u = 0; u < 4; ++u) {
      aF[buf][u].q[0] = *(const float4*)(arow[u] + q0 + sel * 16);
      aF[buf][u].q[1] = *(const float4*)(arow[u] + q0 + sel * 16 + 8);
    }
  };
  loadK(0, 0);
#pragma unroll
  for (int kk = 0; kk < 16; ++kk) {
    int cur = kk & 1, nxt = cur ^ 1;
    if (kk < 15) loadK(kk + 1, nxt);                // in-flight next K-step
#pragma unroll
    for (int u = 0; u < 4; ++u)
      acc[u] = __builtin_amdgcn_wmma_f32_16x16x32_bf16(false, aF[cur][u].v, false,
                                                       bF[cur].v, (short)0, acc[u],
                                                       false, false);
  }
#pragma unroll
  for (int u = 0; u < 4; ++u) {
    if (mt[u] < 31) {
#pragma unroll
      for (int j = 0; j < 8; ++j) {
        int p = mt[u] * 16 + sel * 8 + j;           // D: VGPR j -> M row j(+8)
        sP[p * T_N + row] = acc[u][j];
      }
    }
  }
  __syncthreads();
  // scores[y,x] = sum_t P[(y+dy-2)*22+(x+dx-2), t]
  float* out = outWS + (size_t)pair * OUT_N;
  for (int idx = tid; idx < OUT_N; idx += blockDim.x) {
    int y = idx / WO, x = idx - y * WO;
    float sum = 0.f;
#pragma unroll
    for (int t = 0; t < 16; ++t) {
      int dy = t >> 2, dx = t & 3;
      int py = y + dy - 2, px = x + dx - 2;
      if ((unsigned)py < H_N && (unsigned)px < W_N)
        sum += sP[(py * W_N + px) * T_N + t];
    }
    out[idx] = sum;
  }
}

// ---------------------------------------------------------------------------
// Softmax over 529 + residual. grid: 96 blocks, 256 threads.
// ---------------------------------------------------------------------------
__global__ void k_softmax(float* __restrict__ scoresWS, const float* __restrict__ labelWS,
                          float* __restrict__ resWS) {
  __shared__ float sRed[32];
  int pair = blockIdx.x, tid = threadIdx.x;
  float* sc = scoresWS + (size_t)pair * OUT_N;
  const float* lab = labelWS + (size_t)pair * OUT_N;
  float* res = resWS + (size_t)pair * OUT_N;
  float m = -3.0e38f;
  for (int idx = tid; idx < OUT_N; idx += blockDim.x) m = fmaxf(m, sc[idx]);
  m = blockReduceMax(m, sRed);
  float part = 0.f;
  for (int idx = tid; idx < OUT_N; idx += blockDim.x) {
    float e = expf(sc[idx] - m);
    sc[idx] = e;
    part += e;
  }
  float sum = blockReduceSum(part, sRed);
  float inv = 1.f / sum;
  for (int idx = tid; idx < OUT_N; idx += blockDim.x) {
    float p = sc[idx] * inv;
    sc[idx] = p;
    res[idx] = (p - lab[idx]) * (1.0f / (float)I_N);
  }
}

// ---------------------------------------------------------------------------
// Precompute shifted residual B-matrix in bf16, tap-major: Rb[pair][t][q],
// Rb = res[qy-dy+2, qx-dx+2] (0 outside). grid: 96 blocks, 256 threads.
// ---------------------------------------------------------------------------
__global__ void k_shift(const float* __restrict__ resWS, bf16* __restrict__ RbWS) {
  __shared__ float sR[OUT_N];
  int pair = blockIdx.x, tid = threadIdx.x;
  const float* res = resWS + (size_t)pair * OUT_N;
  for (int idx = tid; idx < OUT_N; idx += blockDim.x) sR[idx] = res[idx];
  __syncthreads();
  bf16* rb = RbWS + (size_t)pair * FLT_N;
  for (int idx = tid; idx < FLT_N; idx += blockDim.x) {
    int t = idx >> 9, q = idx & 511;
    int dy = t >> 2, dx = t & 3;
    float v = 0.f;
    if (q < P_N) {
      int qy = q / W_N, qx = q - qy * W_N;
      int y = qy - dy + 2, x = qx - dx + 2;
      if ((unsigned)y < HO && (unsigned)x < WO) v = sR[y * WO + x];
    }
    rb[idx] = (bf16)v;
  }
}

// ---------------------------------------------------------------------------
// Transposed conv as WMMA GEMM: fgrad[c,t] = sum_{i,q} feat[c,q]*Rb[q,t], +reg*f.
// grid: 32 blocks = s, 256 threads = 8 waves; each wave owns 4 channel-tiles.
// B (shifted residual) staged in LDS; K-loop double-buffered.
// ---------------------------------------------------------------------------
__global__ void k_fgrad(const bf16* __restrict__ featT, const bf16* __restrict__ RbWS,
                        const float* __restrict__ fW, bf16* __restrict__ gbf,
                        float* __restrict__ gW, const float* __restrict__ scal) {
  __shared__ bf16 sRb[I_N * FLT_N];   // 48 KB
  int s = blockIdx.x, tid = threadIdx.x;
  int wave = tid >> 5, lane = tid & 31;
  int sel = lane >> 4, row = lane & 15;
  for (int i = 0; i < I_N; ++i) {
    const unsigned int* src = (const unsigned int*)(RbWS + (size_t)(i * 32 + s) * FLT_N);
    unsigned int* dst = (unsigned int*)(sRb + i * FLT_N);
    for (int idx = tid; idx < FLT_N / 2; idx += blockDim.x) dst[idx] = src[idx];
  }
  __syncthreads();
  float reg = scal[0];
  int t = row;
  int ct[4];
#pragma unroll
  for (int u = 0; u < 4; ++u) ct[u] = wave + u * 8;   // 32 tiles, all valid

  v8f acc[4] = {};
  for (int i = 0; i < I_N; ++i) {
    const bf16* arow[4];
#pragma unroll
    for (int u = 0; u < 4; ++u)
      arow[u] = featT + ((size_t)(i * 32 + s) * C_N + (ct[u] * 16 + row)) * PP_N;
    const bf16* brow = sRb + i * FLT_N + row * C_N;   // lane column = tap
    BFrag aF[2][4], bF[2];
    auto loadK = [&](int kk, int buf) {
      int q0 = kk * 32;
      const bf16* bp = brow + q0 + sel * 16;
      bF[buf].q[0] = *(const float4*)(bp);
      bF[buf].q[1] = *(const float4*)(bp + 8);
#pragma unroll
      for (int u = 0; u < 4; ++u) {
        aF[buf][u].q[0] = *(const float4*)(arow[u] + q0 + sel * 16);
        aF[buf][u].q[1] = *(const float4*)(arow[u] + q0 + sel * 16 + 8);
      }
    };
    loadK(0, 0);
#pragma unroll
    for (int kk = 0; kk < 16; ++kk) {
      int cur = kk & 1, nxt = cur ^ 1;
      if (kk < 15) loadK(kk + 1, nxt);
#pragma unroll
      for (int u = 0; u < 4; ++u)
        acc[u] = __builtin_amdgcn_wmma_f32_16x16x32_bf16(false, aF[cur][u].v, false,
                                                         bF[cur].v, (short)0, acc[u],
                                                         false, false);
    }
  }
#pragma unroll
  for (int u = 0; u < 4; ++u) {
    union { float4 f4; bf16 e[8]; } pk;
    int cbase = ct[u] * 16 + sel * 8;
#pragma unroll
    for (int j = 0; j < 8; ++j) {
      int c = cbase + j;
      float g = acc[u][j] + reg * fW[(size_t)s * FLT_N + c * T_N + t];
      gW[(size_t)s * FLT_N + c * T_N + t] = g;
      pk.e[j] = (bf16)g;
    }
    *(float4*)(gbf + (size_t)s * FLT_N + t * C_N + cbase) = pk.f4;
  }
}

// ---------------------------------------------------------------------------
// Newton scalars + filter update. grid: 32 blocks = s, 256 threads.
// ---------------------------------------------------------------------------
__global__ void k_update(const float* __restrict__ scoresWS, const float* __restrict__ sgradWS,
                         float* __restrict__ fW, const float* __restrict__ gW,
                         bf16* __restrict__ fbf, const float* __restrict__ scal,
                         float* __restrict__ dOut, int writeOut) {
  __shared__ float sRed[32];
  int s = blockIdx.x, tid = threadIdx.x;
  float reg = scal[0], step = scal[1];
  float ghg = 0.f;
  for (int i = 0; i < I_N; ++i) {
    const float* sc = scoresWS + (size_t)(i * 32 + s) * OUT_N;
    const float* sg = sgradWS + (size_t)(i * 32 + s) * OUT_N;
    float p1 = 0.f, p2 = 0.f;
    for (int idx = tid; idx < OUT_N; idx += blockDim.x) {
      float a = sc[idx], b = sg[idx];
      p1 += a * b;
      p2 += a * b * b;
    }
    float t1 = blockReduceSum(p1, sRed);
    float t2 = blockReduceSum(p2, sRed);
    ghg += fmaxf(t2 - t1 * t1, 0.f);   // sum_p sgrad*hes
  }
  ghg *= (1.0f / (float)I_N);
  float pa = 0.f;
  const float* g = gW + (size_t)s * FLT_N;
  for (int idx = tid; idx < FLT_N; idx += blockDim.x) { float v = g[idx]; pa += v * v; }
  float aNum = blockReduceSum(pa, sRed);
  float aDen = fmaxf(ghg + reg * aNum, 1e-8f);
  float factor = step * (aNum / aDen);
  float* f = fW + (size_t)s * FLT_N;
  for (int idx = tid; idx < FLT_N; idx += blockDim.x) {
    float v = f[idx] - factor * g[idx];
    f[idx] = v;
    int c = idx >> 4, t = idx & 15;
    fbf[(size_t)s * FLT_N + t * C_N + c] = (bf16)v;
    if (writeOut) dOut[(size_t)s * FLT_N + idx] = v;
  }
}

// ---------------------------------------------------------------------------
extern "C" void kernel_launch(void* const* d_in, const int* in_sizes, int n_in,
                              void* d_out, int out_size, void* d_ws, size_t ws_size,
                              hipStream_t stream) {
  const float* filt = (const float*)d_in[0];
  const float* feat = (const float*)d_in[1];
  const float* bb   = (const float*)d_in[2];
  const float* lsl  = (const float*)d_in[3];
  const float* freg = (const float*)d_in[4];

  char* ws = (char*)d_ws;
  size_t off = 0;
  bf16* featP = (bf16*)(ws + off); off += (size_t)96 * PP_N * C_N * 2;  // 46.5 MB
  bf16* featT = (bf16*)(ws + off); off += (size_t)96 * PP_N * C_N * 2;  // 46.5 MB
  float* fW   = (float*)(ws + off); off += (size_t)32 * FLT_N * 4;      // 1 MB
  float* gW   = (float*)(ws + off); off += (size_t)32 * FLT_N * 4;      // 1 MB
  bf16* fbf   = (bf16*)(ws + off); off += (size_t)32 * FLT_N * 2;       // 0.5 MB
  bf16* gbf   = (bf16*)(ws + off); off += (size_t)32 * FLT_N * 2;       // 0.5 MB
  bf16* RbWS  = (bf16*)(ws + off); off += (size_t)96 * FLT_N * 2;       // 1.5 MB
  float* labelWS  = (float*)(ws + off); off += (size_t)96 * OUT_N * 4;
  float* scoresWS = (float*)(ws + off); off += (size_t)96 * OUT_N * 4;
  float* sgradWS  = (float*)(ws + off); off += (size_t)96 * OUT_N * 4;
  float* resWS    = (float*)(ws + off); off += (size_t)96 * OUT_N * 4;
  float* scal     = (float*)(ws + off); off += 256;
  // total ~103 MB

  k_pack<<<96, 256, 0, stream>>>(filt, feat, bb, lsl, freg, fW, fbf, featP, featT,
                                 labelWS, scal);
  for (int it = 0; it < 5; ++it) {
    k_conv<<<96, 256, 0, stream>>>(featP, fbf, scoresWS);
    k_softmax<<<96, 256, 0, stream>>>(scoresWS, labelWS, resWS);
    k_shift<<<96, 256, 0, stream>>>(resWS, RbWS);
    k_fgrad<<<32, 256, 0, stream>>>(featT, RbWS, fW, gbf, gW, scal);
    k_conv<<<96, 256, 0, stream>>>(featP, gbf, sgradWS);
    k_update<<<32, 256, 0, stream>>>(scoresWS, sgradWS, fW, gW, fbf, scal,
                                     (float*)d_out, it == 4 ? 1 : 0);
  }
}
